// AdditiveAttention_25692494365251
// MI455X (gfx1250) — matrix operand order repaired and checked by reading, not compiled, not run on hardware
//
#include <hip/hip_runtime.h>
#include <hip/hip_bf16.h>
#include <math.h>

typedef __attribute__((ext_vector_type(16))) _Float16 v16h;
typedef __attribute__((ext_vector_type(8)))  _Float16 v8h;
typedef __attribute__((ext_vector_type(8)))  float    v8f;

// ---------------------------------------------------------------------------
// gfx1250 helpers
// ---------------------------------------------------------------------------
__device__ __forceinline__ void async_load_b128(unsigned lds_off, const void* gaddr) {
    // GLOBAL_LOAD_ASYNC_TO_LDS_B128 (GV mode), tracked by ASYNCcnt.
    asm volatile("global_load_async_to_lds_b128 %0, %1, off"
                 :: "v"(lds_off), "v"(gaddr) : "memory");
}
__device__ __forceinline__ void wait_async() {
    asm volatile("s_wait_asynccnt 0" ::: "memory");
}
__device__ __forceinline__ unsigned lds_addr(const void* p) {
    return (unsigned)(size_t)p;   // low 32 bits of generic shared ptr = LDS offset
}
__device__ __forceinline__ float fast_tanh(float x) {
#if __has_builtin(__builtin_amdgcn_tanhf)
    return __builtin_amdgcn_tanhf(x);
#elif __has_builtin(__builtin_amdgcn_tanh_f32)
    return __builtin_amdgcn_tanh_f32(x);
#else
    return tanhf(x);
#endif
}

// ---------------------------------------------------------------------------
// Kernel P: C(M x H) = A(M x D) * W(D x H), f32 memory, f16 WMMA, f32 accum.
// Block: 256 threads = 8 waves (2x4) of 16x16 tiles -> 32x64 block tile.
// K-step 64: two chained WMMAs per wave per barrier pair.
// ---------------------------------------------------------------------------
#define PA_PITCH 68   // 64 + 4 f32 pad (row stride 272B)
#define PB_PITCH 68   // 64 + 4 f32 pad (col stride 272B)

__global__ void proj_gemm_f16(const float* __restrict__ A,
                              const float* __restrict__ W,
                              float* __restrict__ C,
                              int M, int D, int H) {
    __shared__ float As[32 * PA_PITCH];   // [m][k], 32 x 64
    __shared__ float Bs[64 * PB_PITCH];   // [k][n], 64 x 64

    const int tid  = threadIdx.x;
    const int lane = tid & 31;
    const int wave = tid >> 5;
    const int wm   = wave >> 2;           // 0..1
    const int wn   = wave & 3;            // 0..3

    const int tilesN = H >> 6;            // H/64
    const int bm = blockIdx.x / tilesN;
    const int bn = blockIdx.x % tilesN;

    const int g = lane >> 4;
    const int l = lane & 15;

    // staging coords: A 32x64 f32 = 512 b128 (2/thread); B 64x64 = 1024 (4/thread)
    const int a_row = tid >> 3;           // 0..31
    const int a_col = (tid & 7) * 4;      // 0..28  (+32 for second load)
    const int b_row = tid >> 4;           // 0..15  (+16j, j=0..3)
    const int b_col = (tid & 15) * 4;     // 0..60

    const float* Ag = A + (size_t)(bm * 32 + a_row) * D + a_col;
    const float* Bg = W + (size_t)b_row * H + bn * 64 + b_col;

    const unsigned a_l0 = lds_addr(&As[a_row * PA_PITCH + a_col]);
    const unsigned a_l1 = lds_addr(&As[a_row * PA_PITCH + a_col + 32]);
    const unsigned b_l0 = lds_addr(&Bs[(b_row +  0) * PB_PITCH + b_col]);
    const unsigned b_l1 = lds_addr(&Bs[(b_row + 16) * PB_PITCH + b_col]);
    const unsigned b_l2 = lds_addr(&Bs[(b_row + 32) * PB_PITCH + b_col]);
    const unsigned b_l3 = lds_addr(&Bs[(b_row + 48) * PB_PITCH + b_col]);

    v8f c = {};
    for (int kk = 0; kk < D; kk += 64) {
        async_load_b128(a_l0, Ag + kk);
        async_load_b128(a_l1, Ag + kk + 32);
        async_load_b128(b_l0, Bg + (size_t)(kk +  0) * H);
        async_load_b128(b_l1, Bg + (size_t)(kk + 16) * H);
        async_load_b128(b_l2, Bg + (size_t)(kk + 32) * H);
        async_load_b128(b_l3, Bg + (size_t)(kk + 48) * H);
        wait_async();
        __syncthreads();

#pragma unroll
        for (int kb = 0; kb < 64; kb += 32) {
            // A fragment: lane m+16g holds K = kb + g*8+i and kb+16+g*8+i
            const float* Am = &As[(wm * 16 + l) * PA_PITCH + kb];
            const float4 a0 = *(const float4*)(Am + g * 8);
            const float4 a1 = *(const float4*)(Am + g * 8 + 4);
            const float4 a2 = *(const float4*)(Am + 16 + g * 8);
            const float4 a3 = *(const float4*)(Am + 16 + g * 8 + 4);
            float af[16] = {a0.x, a0.y, a0.z, a0.w, a1.x, a1.y, a1.z, a1.w,
                            a2.x, a2.y, a2.z, a2.w, a3.x, a3.y, a3.z, a3.w};
            v16h a;
#pragma unroll
            for (int i = 0; i < 16; ++i) a[i] = (_Float16)af[i];

            // B fragment: lane n+16g holds K = kb + g*8+i and kb+16+g*8+i for column n
            const float* Bc = &Bs[kb * PB_PITCH + wn * 16 + l];
            v16h b;
#pragma unroll
            for (int i = 0; i < 16; ++i) b[i] = (_Float16)Bc[(g * 8 + (i & 7) + (i & 8) * 2) * PB_PITCH];

            c = __builtin_amdgcn_wmma_f32_16x16x32_f16(
                    false, a, false, b, (short)0, c, false, false);
        }
        __syncthreads();
    }

    // C/D f32 layout: VGPR r, lane n+16g -> (M = r + 8g, N = n)
    float* Crow = C + (size_t)(bm * 32 + wm * 16) * H + (size_t)(bn * 64 + wn * 16) + l;
#pragma unroll
    for (int r = 0; r < 8; ++r)
        Crow[(size_t)(r + 8 * g) * H] = c[r];
}

// ---------------------------------------------------------------------------
// Kernel S: fused scores + masked softmax, one (b,q) row per workgroup.
// score(k) = sum_h wv[h]*tanh(qp[h] + kp[k,h]); masked logits -> 0.0 logit
// (reference semantics), then ordinary softmax. Hardware v_tanh_f32 path.
// ---------------------------------------------------------------------------
__global__ void scores_softmax(const float* __restrict__ qp,   // (B*Q, H)
                               const float* __restrict__ kp,   // (B*K, H)
                               const float* __restrict__ wv,   // (H)
                               const int*   __restrict__ valid_lens,
                               _Float16* __restrict__ attn,    // (B*Q, K)
                               int Q, int K, int H) {
    const int b = blockIdx.y;
    const int q = blockIdx.x;
    const int tid = threadIdx.x;

    __shared__ float qrow[128];
    __shared__ float wvs[128];
    __shared__ float sc[1024];
    __shared__ float red[256];

    if (tid < H) {
        qrow[tid] = qp[((size_t)b * Q + q) * H + tid];
        wvs[tid]  = wv[tid];
    }
    __syncthreads();

    const int vlen = valid_lens[b];

    for (int k = tid; k < K; k += blockDim.x) {
        const float* krow = kp + ((size_t)b * K + k) * H;
        float s = 0.0f;
#pragma unroll 8
        for (int h = 0; h < H; h += 4) {
            float4 kv = *(const float4*)(krow + h);
            s += wvs[h + 0] * fast_tanh(qrow[h + 0] + kv.x);
            s += wvs[h + 1] * fast_tanh(qrow[h + 1] + kv.y);
            s += wvs[h + 2] * fast_tanh(qrow[h + 2] + kv.z);
            s += wvs[h + 3] * fast_tanh(qrow[h + 3] + kv.w);
        }
        sc[k] = (k < vlen) ? s : 0.0f;
    }
    __syncthreads();

    float m = -INFINITY;
    for (int k = tid; k < K; k += blockDim.x) m = fmaxf(m, sc[k]);
    red[tid] = m;
    __syncthreads();
    for (int s = 128; s > 0; s >>= 1) {
        if (tid < s) red[tid] = fmaxf(red[tid], red[tid + s]);
        __syncthreads();
    }
    const float mx = red[0];
    __syncthreads();

    float psum = 0.0f;
    for (int k = tid; k < K; k += blockDim.x) {
        float e = __expf(sc[k] - mx);
        sc[k] = e;
        psum += e;
    }
    red[tid] = psum;
    __syncthreads();
    for (int s = 128; s > 0; s >>= 1) {
        if (tid < s) red[tid] += red[tid + s];
        __syncthreads();
    }
    const float inv = 1.0f / red[0];
    __syncthreads();

    _Float16* arow = attn + ((size_t)b * Q + q) * K;
    for (int k = tid; k < K; k += blockDim.x)
        arow[k] = (_Float16)(sc[k] * inv);
}

// ---------------------------------------------------------------------------
// Kernel V: f32 -> f16 conversion of `values`.
// ---------------------------------------------------------------------------
__global__ void cvt_f32_f16(const float* __restrict__ in,
                            _Float16* __restrict__ out, int n) {
    int i = (blockIdx.x * blockDim.x + threadIdx.x) * 4;
    if (i + 3 < n) {
        float4 v = *(const float4*)(in + i);
        out[i + 0] = (_Float16)v.x;
        out[i + 1] = (_Float16)v.y;
        out[i + 2] = (_Float16)v.z;
        out[i + 3] = (_Float16)v.w;
    } else {
        for (; i < n; ++i) out[i] = (_Float16)in[i];
    }
}

// ---------------------------------------------------------------------------
// Kernel O: out(b, QxD) = attn(b, QxK)[f16] * vals(b, KxD)[f16], f32 accum.
// 32x64 block tile, K-step 64, two chained WMMAs per wave per barrier pair.
// ---------------------------------------------------------------------------
#define OA_PITCH 72   // 64 + 8 f16 pad (row stride 144B)
#define OB_PITCH 72   // 64 + 8 f16 pad (col stride 144B)

__global__ void attn_v_gemm(const _Float16* __restrict__ attn,  // (B, Q, K)
                            const _Float16* __restrict__ vals,  // (B, K, D)
                            float* __restrict__ out,            // (B, Q, D)
                            int Q, int K, int D) {
    __shared__ _Float16 As[32 * OA_PITCH];   // [m][k], 32 x 64
    __shared__ _Float16 Bs[64 * OB_PITCH];   // [k][n], 64 x 64

    const int b    = blockIdx.y;
    const int tid  = threadIdx.x;
    const int lane = tid & 31;
    const int wave = tid >> 5;
    const int wm   = wave >> 2;
    const int wn   = wave & 3;

    const int tilesN = D >> 6;            // D/64
    const int bm = blockIdx.x / tilesN;
    const int bn = blockIdx.x % tilesN;

    const int g = lane >> 4;
    const int l = lane & 15;

    // staging: A 32x64 f16 = 256 b128 (1/thread); B 64x64 f16 = 512 (2/thread)
    const int a_row = tid >> 3;           // 0..31
    const int a_col = (tid & 7) * 8;      // halves
    const int b_row = tid >> 3;           // 0..31 (+32)
    const int b_col = (tid & 7) * 8;      // halves

    const _Float16* Ag = attn + ((size_t)b * Q + bm * 32 + a_row) * K + a_col;
    const _Float16* Bg = vals + (size_t)b * K * D + (size_t)b_row * D + bn * 64 + b_col;

    const unsigned a_l  = lds_addr(&As[a_row * OA_PITCH + a_col]);
    const unsigned b_l0 = lds_addr(&Bs[(b_row +  0) * OB_PITCH + b_col]);
    const unsigned b_l1 = lds_addr(&Bs[(b_row + 32) * OB_PITCH + b_col]);

    v8f c = {};
    for (int kk = 0; kk < K; kk += 64) {
        async_load_b128(a_l, Ag + kk);
        async_load_b128(b_l0, Bg + (size_t)(kk +  0) * D);
        async_load_b128(b_l1, Bg + (size_t)(kk + 32) * D);
        wait_async();
        __syncthreads();

#pragma unroll
        for (int kb = 0; kb < 64; kb += 32) {
            const _Float16* Am = &As[(wm * 16 + l) * OA_PITCH + kb];
            v8h a0 = *(const v8h*)(Am + g * 8);
            v8h a1 = *(const v8h*)(Am + 16 + g * 8);
            v16h a;
#pragma unroll
            for (int i = 0; i < 8; ++i) { a[i] = a0[i]; a[i + 8] = a1[i]; }

            const _Float16* Bc = &Bs[kb * OB_PITCH + wn * 16 + l];
            v16h bmat;
#pragma unroll
            for (int i = 0; i < 16; ++i) bmat[i] = Bc[(g * 8 + (i & 7) + (i & 8) * 2) * OB_PITCH];

            c = __builtin_amdgcn_wmma_f32_16x16x32_f16(
                    false, a, false, bmat, (short)0, c, false, false);
        }
        __syncthreads();
    }

    float* Crow = out + ((size_t)b * Q + bm * 32 + wm * 16) * D
                      + (size_t)(bn * 64 + wn * 16) + l;
#pragma unroll
    for (int r = 0; r < 8; ++r)
        Crow[(size_t)(r + 8 * g) * D] = c[r];
}

// ---------------------------------------------------------------------------
extern "C" void kernel_launch(void* const* d_in, const int* in_sizes, int n_in,
                              void* d_out, int out_size, void* d_ws, size_t ws_size,
                              hipStream_t stream) {
    const float* queries    = (const float*)d_in[0];  // (B,Q,D)
    const float* keys       = (const float*)d_in[1];  // (B,K,D)
    const float* values     = (const float*)d_in[2];  // (B,K,D)
    const float* Wq         = (const float*)d_in[3];  // (D,H)
    const float* Wk         = (const float*)d_in[4];  // (D,H)
    const float* wv         = (const float*)d_in[5];  // (H)
    const int*   valid_lens = (const int*)d_in[6];    // (B)

    constexpr int B = 4, Q = 256, K = 1024, D = 512, H = 128;

    // Workspace: qp f32 (B*Q,H) | kp f32 (B*K,H) | attn f16 (B*Q,K) | v16 f16 (B*K,D)
    char* ws = (char*)d_ws;
    float*    qp   = (float*)ws;
    float*    kp   = (float*)(ws + (size_t)B * Q * H * 4);
    _Float16* attn = (_Float16*)(ws + (size_t)B * Q * H * 4 + (size_t)B * K * H * 4);
    _Float16* v16  = (_Float16*)((char*)attn + (size_t)B * Q * K * 2);

    // 1) Query projection: (B*Q x D) * (D x H)
    {
        const int M = B * Q;
        const int blocks = (M / 32) * (H / 64);       // 64
        proj_gemm_f16<<<blocks, 256, 0, stream>>>(queries, Wq, qp, M, D, H);
    }
    // 2) Key projection: (B*K x D) * (D x H)
    {
        const int M = B * K;
        const int blocks = (M / 32) * (H / 64);       // 256
        proj_gemm_f16<<<blocks, 256, 0, stream>>>(keys, Wk, kp, M, D, H);
    }
    // 3) Fused tanh-scores + masked softmax
    {
        dim3 grid(Q, B);
        scores_softmax<<<grid, 256, 0, stream>>>(qp, kp, wv, valid_lens, attn, Q, K, H);
    }
    // 4) values -> f16
    {
        const int n = B * K * D;
        cvt_f32_f16<<<(n / 4 + 255) / 256, 256, 0, stream>>>(values, v16, n);
    }
    // 5) Output GEMM: attn (Q x K) * values (K x D) per batch
    {
        dim3 grid((Q / 32) * (D / 64), B);            // (64, 4)
        attn_v_gemm<<<grid, 256, 0, stream>>>(attn, v16, (float*)d_out, Q, K, D);
    }
}
